// CGCNNTorchModel_12575664242924
// MI455X (gfx1250) — matrix operand order repaired and checked by reading, not compiled, not run on hardware
//
#include <hip/hip_runtime.h>
#include <hip/hip_bf16.h>
#include <math.h>

#define N_NODES 100000
#define N_EDGES 1600000
#define NGR     512
#define NODE_DIM 92
#define EDGE_DIM 41
#define HD      64
#define KDIM    169     // 2H + EDGE_DIM
#define KPAD    192     // padded to 6 * 32
#define NCH     128     // 2H
#define PRED_H  128
#define NTILES  (N_EDGES / 16)     // 100000
#define NSUPER  (NTILES / 8)       // 12500
#define PACK_PER_LAYER (8 * 6 * 32 * 16)   // 24576 bf16 values

typedef __attribute__((ext_vector_type(16))) __bf16 v16bf;
typedef __attribute__((ext_vector_type(8)))  float  v8f;

struct U16x8 { unsigned int q[4]; };               // 16 bytes, trivial
union Frag { v16bf v; U16x8 h[2]; };               // 32 bytes

struct F4 { float x, y, z, w; };                   // trivial 16B vector load

__device__ __forceinline__ unsigned short f2bf(float f) {
    unsigned int u = __float_as_uint(f);
    unsigned int r = u + 0x7FFFu + ((u >> 16) & 1u);   // round-to-nearest-even
    return (unsigned short)(r >> 16);
}
__device__ __forceinline__ unsigned int pack2(float a, float b) {
    return (unsigned int)f2bf(a) | ((unsigned int)f2bf(b) << 16);
}
__device__ __forceinline__ float sigm(float x)  { return 1.f / (1.f + __expf(-x)); }
__device__ __forceinline__ float splus(float x) { return (x > 20.f) ? x : log1pf(__expf(x)); }

// ---------------------------------------------------------------------------
// Pack conv_W[l] (fp32 [169][128]) into per-lane bf16 B-fragments:
// layout [layer][frag = nt*6+ks][lane 0..31][elem 0..15]
// lane<16 : n = nt*16+lane,      k = ks*32 + (elem&7) + 16*(elem>>3)      (K 0..7, 16..23)
// lane>=16: n = nt*16+(lane-16), k = ks*32 + 8 + (elem&7) + 16*(elem>>3)  (K 8..15, 24..31)
// ---------------------------------------------------------------------------
__global__ void pack_weights(const float* __restrict__ W, unsigned short* __restrict__ out) {
    int idx = blockIdx.x * blockDim.x + threadIdx.x;
    if (idx >= 3 * PACK_PER_LAYER) return;
    int l = idx / PACK_PER_LAYER, r = idx % PACK_PER_LAYER;
    int frag = r >> 9;            // /512
    int within = r & 511;
    int lane = within >> 4, elem = within & 15;
    int nt = frag / 6, ks = frag % 6;
    int n  = nt * 16 + (lane & 15);
    int kl = ((lane >> 4) << 3) + (elem & 7) + ((elem >> 3) << 4);
    int k  = ks * 32 + kl;
    float v = (k < KDIM) ? W[(size_t)l * KDIM * NCH + (size_t)k * NCH + n] : 0.f;
    out[idx] = f2bf(v);
}

// ---------------------------------------------------------------------------
// Node embedding: h = node_feats @ emb_W + emb_b   (memory-bound, VALU)
// ---------------------------------------------------------------------------
__global__ __launch_bounds__(256) void embed_kernel(
        const float* __restrict__ nf, const float* __restrict__ W,
        const float* __restrict__ b,  float* __restrict__ h) {
    __shared__ float Wl[NODE_DIM * HD];
    __shared__ float rows[4][NODE_DIM];
    int tid = threadIdx.x;
    for (int i = tid; i < NODE_DIM * HD; i += 256) Wl[i] = W[i];
    __syncthreads();
    int nl = tid >> 6, j = tid & 63;
    for (int g = blockIdx.x; g < N_NODES / 4; g += gridDim.x) {
        int base = g * 4;
        for (int i = tid; i < 4 * NODE_DIM; i += 256)
            rows[i / NODE_DIM][i % NODE_DIM] = nf[(size_t)base * NODE_DIM + i];
        __syncthreads();
        float acc = b[j];
        for (int k = 0; k < NODE_DIM; ++k) acc += rows[nl][k] * Wl[k * HD + j];
        h[(size_t)(base + nl) * HD + j] = acc;
        __syncthreads();
    }
}

// ---------------------------------------------------------------------------
// Conv edge-GEMM pass. pass==0: accumulate per-channel sum/sumsq of C
// (bias cancels under BN). pass==1: z = C*scale+shift; msg = sig(z_lo)*sp(z_hi);
// scatter-add msg into h_acc[src].
// Block = 256 threads = 8 waves; block stages 8 edge-tiles of A into LDS
// (h gathered with b128 float4 loads), wave w computes tile w across all 8
// n-tiles with v_wmma_f32_16x16x32_bf16; B fragments are register-resident
// (compiler hoists the 48 fragment loads into the extended VGPR file).
// ---------------------------------------------------------------------------
__global__ __launch_bounds__(256) void conv_pass(
        const float* __restrict__ h_in, float* __restrict__ h_acc,
        const int* __restrict__ ei, const float* __restrict__ ef,
        const unsigned short* __restrict__ packB,
        float* __restrict__ stSum, float* __restrict__ stSq,
        const float* __restrict__ scale, const float* __restrict__ shift,
        int pass) {
    __shared__ __align__(16) unsigned short At[8][16 * KPAD];   // 48 KB
    __shared__ int   sSrc[8][16];
    __shared__ float bSum[NCH], bSq[NCH];

    const int tid  = threadIdx.x;
    const int w    = tid >> 5;
    const int lane = tid & 31;
    const int m    = lane & 15;
    const int kb   = (lane >> 4) << 3;   // A-frag K sub-base: 0 or 8

    if (pass == 0 && tid < NCH) { bSum[tid] = 0.f; bSq[tid] = 0.f; }

    // Staging role (fixed per thread): 2 threads per row, 8 tiles x 16 rows.
    const int srow = tid >> 1, shalf = tid & 1;
    const int stl = srow >> 4, sr = srow & 15;
    unsigned short* arow = &At[stl][sr * KPAD];

    // Zero the constant K-padding (cols 170..191) once; it stays zero.
    if (shalf == 1) {
        #pragma unroll
        for (int c = 170; c < KPAD; c += 2) *(unsigned int*)&arow[c] = 0u;
    }

    for (int s = blockIdx.x; s < NSUPER; s += gridDim.x) {
        // ---- stage A: vectorized gather ----
        {
            int e = (s * 8 + stl) * 16 + sr;
            int src = ei[e];
            int dst = ei[N_EDGES + e];
            const float* fe = ef + (size_t)e * EDGE_DIM;
            if (shalf == 0) {
                sSrc[stl][sr] = src;
                const F4* hs4 = (const F4*)(h_in + (size_t)src * HD);
                const F4* hd4 = (const F4*)(h_in + (size_t)dst * HD);
                #pragma unroll
                for (int q = 0; q < 16; ++q) {          // cols 0..63 = h[src]
                    F4 v = hs4[q];
                    *(unsigned int*)&arow[q * 4]     = pack2(v.x, v.y);
                    *(unsigned int*)&arow[q * 4 + 2] = pack2(v.z, v.w);
                }
                #pragma unroll
                for (int q = 0; q < 8; ++q) {           // cols 64..95 = h[dst][0..31]
                    F4 v = hd4[q];
                    *(unsigned int*)&arow[64 + q * 4]     = pack2(v.x, v.y);
                    *(unsigned int*)&arow[64 + q * 4 + 2] = pack2(v.z, v.w);
                }
            } else {
                const F4* hd4 = (const F4*)(h_in + (size_t)dst * HD);
                #pragma unroll
                for (int q = 8; q < 16; ++q) {          // cols 96..127 = h[dst][32..63]
                    F4 v = hd4[q];
                    *(unsigned int*)&arow[64 + q * 4]     = pack2(v.x, v.y);
                    *(unsigned int*)&arow[64 + q * 4 + 2] = pack2(v.z, v.w);
                }
                #pragma unroll
                for (int q = 0; q < 20; ++q) {          // cols 128..167 = edge_feats[0..39]
                    *(unsigned int*)&arow[128 + q * 2] = pack2(fe[q * 2], fe[q * 2 + 1]);
                }
                *(unsigned int*)&arow[168] = pack2(fe[40], 0.f);  // col 168 + pad
            }
        }
        __syncthreads();

        // ---- WMMA: wave w owns tile w, all 8 n-tiles ----
        v8f C[8];
        #pragma unroll
        for (int i = 0; i < 8; ++i) C[i] = (v8f){0, 0, 0, 0, 0, 0, 0, 0};

        const unsigned short* afrag = &At[w][m * KPAD];
        for (int ks = 0; ks < 6; ++ks) {
            Frag a;
            const unsigned short* ap = afrag + ks * 32 + kb;
            a.h[0] = *(const U16x8*)(ap);        // K = kb+0..7
            a.h[1] = *(const U16x8*)(ap + 16);   // K = kb+16..23
            #pragma unroll
            for (int nt = 0; nt < 8; ++nt) {
                Frag b;
                const unsigned short* bp = packB + ((nt * 6 + ks) * 32 + lane) * 16;
                b.h[0] = *(const U16x8*)(bp);
                b.h[1] = *(const U16x8*)(bp + 8);
                C[nt] = __builtin_amdgcn_wmma_f32_16x16x32_bf16(
                            false, a.v, false, b.v, (short)0, C[nt], false, false);
            }
        }

        if (pass == 0) {
            #pragma unroll
            for (int nt = 0; nt < 8; ++nt) {
                float s1 = 0.f, s2 = 0.f;
                #pragma unroll
                for (int v = 0; v < 8; ++v) { float z = C[nt][v]; s1 += z; s2 += z * z; }
                s1 += __shfl_xor(s1, 16, 32);
                s2 += __shfl_xor(s2, 16, 32);
                if (lane < 16) {
                    atomicAdd(&bSum[nt * 16 + lane], s1);
                    atomicAdd(&bSq [nt * 16 + lane], s2);
                }
            }
        } else {
            #pragma unroll
            for (int nt = 0; nt < 8; ++nt) {
                int n = nt * 16 + m;
                float sc = scale[n], sh = shift[n];
                #pragma unroll
                for (int v = 0; v < 8; ++v) C[nt][v] = C[nt][v] * sc + sh;
            }
            #pragma unroll
            for (int nt = 0; nt < 4; ++nt) {
                #pragma unroll
                for (int v = 0; v < 8; ++v) {
                    float g = sigm(C[nt][v]) * splus(C[nt + 4][v]);
                    int r = (lane < 16) ? v : v + 8;
                    int src = sSrc[w][r];
                    atomicAdd(&h_acc[(size_t)src * HD + nt * 16 + m], g);
                }
            }
        }
        __syncthreads();
    }

    if (pass == 0 && tid < NCH) {
        atomicAdd(&stSum[tid], bSum[tid]);
        atomicAdd(&stSq [tid], bSq [tid]);
    }
}

// ---------------------------------------------------------------------------
__global__ void bn_finalize(const float* __restrict__ stSum, const float* __restrict__ stSq,
                            const float* __restrict__ gamma, const float* __restrict__ beta,
                            float* __restrict__ scale, float* __restrict__ shift) {
    int c = threadIdx.x;   // 128 threads
    float mean = stSum[c] / (float)N_EDGES;
    float var  = fmaxf(stSq[c] / (float)N_EDGES - mean * mean, 0.f);
    float inv  = rsqrtf(var + 1e-5f);
    float sc   = gamma[c] * inv;
    scale[c] = sc;
    shift[c] = beta[c] - mean * sc;      // conv bias cancels under BN
}

__global__ void zero_f32(float* p, int n) {
    int i = blockIdx.x * blockDim.x + threadIdx.x;
    if (i < n) p[i] = 0.f;
}

// ---------------------------------------------------------------------------
__global__ void pool_kernel(const float* __restrict__ h, const int* __restrict__ gid,
                            float* __restrict__ psum, float* __restrict__ pcnt) {
    int i = blockIdx.x * blockDim.x + threadIdx.x;
    int n = i >> 6, j = i & 63;
    if (n >= N_NODES) return;
    int g = gid[n];
    atomicAdd(&psum[(size_t)g * HD + j], h[(size_t)n * HD + j]);
    if (j == 0) atomicAdd(&pcnt[g], 1.f);
}

__global__ __launch_bounds__(128) void head_kernel(
        const float* __restrict__ psum, const float* __restrict__ pcnt,
        const float* __restrict__ fcW, const float* __restrict__ fcb,
        const float* __restrict__ oW,  const float* __restrict__ ob,
        float* __restrict__ out) {
    __shared__ float pooled[HD];
    __shared__ float red[PRED_H];
    int g = blockIdx.x, tid = threadIdx.x;
    if (tid < HD) {
        float c = fmaxf(pcnt[g], 1.f);
        pooled[tid] = psum[(size_t)g * HD + tid] / c;
    }
    __syncthreads();
    float acc = fcb[tid];
    for (int k = 0; k < HD; ++k) acc += pooled[k] * fcW[k * PRED_H + tid];
    red[tid] = splus(acc) * oW[tid];
    __syncthreads();
    for (int off = PRED_H / 2; off > 0; off >>= 1) {
        if (tid < off) red[tid] += red[tid + off];
        __syncthreads();
    }
    if (tid == 0) out[g] = red[0] + ob[0];
}

// ---------------------------------------------------------------------------
extern "C" void kernel_launch(void* const* d_in, const int* in_sizes, int n_in,
                              void* d_out, int out_size, void* d_ws, size_t ws_size,
                              hipStream_t stream) {
    const float* nf   = (const float*)d_in[0];
    const int*   ei   = (const int*)  d_in[1];
    const float* ef   = (const float*)d_in[2];
    const int*   gid  = (const int*)  d_in[3];
    const float* embW = (const float*)d_in[4];
    const float* embB = (const float*)d_in[5];
    const float* convW= (const float*)d_in[6];
    // d_in[7] = conv_b: constant bias cancels under BatchNorm -> unused
    const float* bng  = (const float*)d_in[8];
    const float* bnb  = (const float*)d_in[9];
    const float* fcW  = (const float*)d_in[10];
    const float* fcb  = (const float*)d_in[11];
    const float* oW   = (const float*)d_in[12];
    const float* ob   = (const float*)d_in[13];
    float* out = (float*)d_out;

    // workspace carve (256B aligned)
    char* p = (char*)d_ws;
    auto carve = [&](size_t bytes) -> char* {
        char* r = p; p += (bytes + 255) & ~(size_t)255; return r;
    };
    float*          hA    = (float*)carve((size_t)N_NODES * HD * 4);
    float*          hB    = (float*)carve((size_t)N_NODES * HD * 4);
    unsigned short* packB = (unsigned short*)carve((size_t)3 * PACK_PER_LAYER * 2);
    float*          stats = (float*)carve(256 * 4);         // sum[128] | sumsq[128]
    float*          scl   = (float*)carve(NCH * 4);
    float*          shf   = (float*)carve(NCH * 4);
    float*          pool  = (float*)carve((size_t)(NGR * HD + NGR) * 4);
    float* stSum = stats, *stSq = stats + NCH;
    float* psum = pool, *pcnt = pool + NGR * HD;

    pack_weights<<<(3 * PACK_PER_LAYER + 255) / 256, 256, 0, stream>>>(convW, packB);
    embed_kernel<<<2048, 256, 0, stream>>>(nf, embW, embB, hA);

    float* cur = hA; float* nxt = hB;
    for (int l = 0; l < 3; ++l) {
        const unsigned short* pB = packB + (size_t)l * PACK_PER_LAYER;
        zero_f32<<<1, 256, 0, stream>>>(stats, 256);
        conv_pass<<<1024, 256, 0, stream>>>(cur, nxt, ei, ef, pB, stSum, stSq, scl, shf, 0);
        bn_finalize<<<1, NCH, 0, stream>>>(stSum, stSq, bng + l * NCH, bnb + l * NCH, scl, shf);
        hipMemcpyAsync(nxt, cur, (size_t)N_NODES * HD * 4, hipMemcpyDeviceToDevice, stream);
        conv_pass<<<1024, 256, 0, stream>>>(cur, nxt, ei, ef, pB, stSum, stSq, scl, shf, 1);
        float* t = cur; cur = nxt; nxt = t;
    }

    zero_f32<<<(NGR * HD + NGR + 255) / 256, 256, 0, stream>>>(pool, NGR * HD + NGR);
    pool_kernel<<<(N_NODES * HD) / 256, 256, 0, stream>>>(cur, gid, psum, pcnt);
    head_kernel<<<NGR, PRED_H, 0, stream>>>(psum, pcnt, fcW, fcb, oW, ob, out);
}
